// GCNConvNet_7292854468802
// MI455X (gfx1250) — matrix operand order repaired and checked by reading, not compiled, not run on hardware
//
#include <hip/hip_runtime.h>
#include <hip/hip_bf16.h>

// ---------------------------------------------------------------------------
// GraphSAGE (max aggr) x3 + MLP head for MI455X (gfx1250, wave32, WMMA, TDM).
//
//  - agg scatter: float-atomic-max (int/uint ordering trick), L2-resident
//  - all activations/weights carried in f16; GEMMs use v_wmma_f32_16x16x32_f16
//  - weight tiles staged into LDS by the Tensor Data Mover
//    (tensor_load_to_lds + s_wait_tensorcnt), A tiles staged cooperatively,
//    fragments then come from LDS (ds_load) instead of redundant global reads
// ---------------------------------------------------------------------------

typedef __attribute__((ext_vector_type(16))) _Float16 v16h;
typedef __attribute__((ext_vector_type(8)))  _Float16 v8h;
typedef __attribute__((ext_vector_type(4)))  _Float16 v4h;
typedef __attribute__((ext_vector_type(8)))  float    v8f;

#define FDIM 128
#define NEGINF_BITS 0xFF800000u

#if defined(__has_builtin)
#if __has_builtin(__builtin_amdgcn_tensor_load_to_lds)
#define USE_TDM 1
#endif
#endif

// ---- float atomic max via int/uint ordering trick (agg init to -inf) -------
__device__ __forceinline__ void atomic_max_float(float* addr, float val) {
  if (val >= 0.0f) {
    atomicMax(reinterpret_cast<int*>(addr), __float_as_int(val));
  } else {
    atomicMin(reinterpret_cast<unsigned int*>(addr), __float_as_uint(val));
  }
}

// ---------------------------------------------------------------------------
__global__ void fill_neginf_kernel(unsigned int* __restrict__ p, int n) {
  int i = blockIdx.x * blockDim.x + threadIdx.x;
  if (i < n) p[i] = NEGINF_BITS;
}

// f32 -> f16 pack (weights)
__global__ void pack_w16_kernel(const float* __restrict__ in,
                                _Float16* __restrict__ out, int n) {
  int i = blockIdx.x * blockDim.x + threadIdx.x;
  if (i < n) out[i] = (_Float16)in[i];
}

// agg f32 -> f16, with -inf (empty segment) -> 0
__global__ void pack_agg16_kernel(const float* __restrict__ in,
                                  _Float16* __restrict__ out, int n) {
  int i = blockIdx.x * blockDim.x + threadIdx.x;
  if (i < n) {
    float v = in[i];
    if (__float_as_uint(v) == NEGINF_BITS) v = 0.0f;
    out[i] = (_Float16)v;
  }
}

// ---- scatter max, F = 3 (layer 1 input features, f32 source) ---------------
__global__ void scatter_max3_kernel(const float* __restrict__ x,
                                    const int* __restrict__ src,
                                    const int* __restrict__ dst,
                                    float* __restrict__ agg, int E) {
  int e = blockIdx.x * blockDim.x + threadIdx.x;
  if (e >= E) return;
  int s = src[e], d = dst[e];
  const float* xv = x + (size_t)s * 3;
  float* av = agg + (size_t)d * 3;
#pragma unroll
  for (int f = 0; f < 3; ++f) atomic_max_float(av + f, xv[f]);
}

// ---- scatter max, F = 128: 32 lanes/edge, 4 f16/lane (b64 gather) ----------
__global__ void scatter_max128_kernel(const _Float16* __restrict__ h,
                                      const int* __restrict__ src,
                                      const int* __restrict__ dst,
                                      float* __restrict__ agg, int E) {
  int t = blockIdx.x * blockDim.x + threadIdx.x;
  int e = t >> 5;             // edge index
  int c = (t & 31) << 2;      // feature base (0..124, step 4)
  if (e >= E) return;
  int s = src[e], d = dst[e];
  if (e + 1 < E) {
    // hint the next edge's source row toward cache (global_prefetch_b8)
    __builtin_prefetch(&h[(size_t)src[e + 1] * FDIM + c], 0, 0);
  }
  const v4h v = *reinterpret_cast<const v4h*>(h + (size_t)s * FDIM + c);
  float* a = agg + (size_t)d * FDIM + c;
#pragma unroll
  for (int f = 0; f < 4; ++f) atomic_max_float(a + f, (float)v[f]);
}

// ---- layer 1: 3 -> 128, scalar (K=3 too small for WMMA) --------------------
__global__ void layer1_kernel(const float* __restrict__ x,
                              const float* __restrict__ agg3,
                              const float* __restrict__ W1l,
                              const float* __restrict__ b1l,
                              const float* __restrict__ W1r,
                              _Float16* __restrict__ out, int N) {
  int i = blockIdx.x * blockDim.x + threadIdx.x;
  if (i >= N * FDIM) return;
  int n = i >> 7;
  int o = i & (FDIM - 1);
  const float* xv = x + (size_t)n * 3;
  const float* wl = W1l + o * 3;
  const float* wr = W1r + o * 3;
  float s = b1l[o];
#pragma unroll
  for (int k = 0; k < 3; ++k) {
    float a = agg3[(size_t)n * 3 + k];
    if (__float_as_uint(a) == NEGINF_BITS) a = 0.0f;  // empty-segment fill
    s += a * wl[k] + xv[k] * wr[k];
  }
  out[i] = (_Float16)s;
}

// ---- TDM: DMA one [rows x 128] f16 matrix from global into LDS -------------
#ifdef USE_TDM
__device__ __forceinline__ void tdm_load_w(const _Float16* gptr,
                                           unsigned lds_off, int rows) {
  typedef __attribute__((ext_vector_type(4))) unsigned int u32x4;
  typedef __attribute__((ext_vector_type(8))) int i32x8;
  typedef __attribute__((ext_vector_type(4))) int i32x4;
  unsigned long long ga = (unsigned long long)(__SIZE_TYPE__)gptr;
  u32x4 g0;
  g0[0] = 1u;                                    // count=1, user descriptor
  g0[1] = lds_off;                               // LDS byte address
  g0[2] = (unsigned)(ga & 0xFFFFFFFFu);          // global_addr[31:0]
  g0[3] = (unsigned)(ga >> 32) | (2u << 30);     // global_addr[56:32] | type=2
  i32x8 g1;
  g1[0] = 0x00010000;                 // wg_mask=0, data_size=1 (2 bytes)
  g1[1] = (int)((FDIM & 0xFFFF) << 16);   // tensor_dim0[15:0]=128
  g1[2] = (int)((rows & 0xFFFF) << 16);   // dim0 hi=0 | tensor_dim1[15:0]=rows
  g1[3] = (int)((FDIM & 0xFFFF) << 16);   // dim1 hi=0 | tile_dim0=128
  g1[4] = rows & 0xFFFF;                  // tile_dim1=rows, tile_dim2=0
  g1[5] = FDIM;                           // tensor_dim0_stride = 128
  g1[6] = 0;
  g1[7] = 0;
  i32x4 z4 = {0, 0, 0, 0};                // groups 2/3 unused (2-D tensor)
#if defined(__clang_major__) && (__clang_major__ >= 23)
  i32x8 z8 = {0, 0, 0, 0, 0, 0, 0, 0};
  __builtin_amdgcn_tensor_load_to_lds(g0, g1, z4, z4, z8, 0);
#else
  __builtin_amdgcn_tensor_load_to_lds(g0, g1, z4, z4, 0);
#endif
}
#endif

// ---- WMMA fragment loaders from LDS (f16) ----------------------------------
// A: 16x32 f16 (ISA 7.12.2): lanes 0-15 row M=lane hold K {0..7,16..23};
//    lanes 16-31 row M=lane-16 hold K {8..15,24..31}.
__device__ __forceinline__ v16h load_frag_a_lds(const _Float16* sA, int kb, int lane) {
  const int m   = lane & 15;
  const int hi8 = (lane & 16) >> 1;
  const _Float16* p = sA + m * FDIM + kb + hi8;
  v8h lo = *reinterpret_cast<const v8h*>(p);        // K 0..7   (+hi8)
  v8h hh = *reinterpret_cast<const v8h*>(p + 16);   // K 16..23 (+hi8)
  return __builtin_shufflevector(lo, hh, 0, 1, 2, 3, 4, 5, 6, 7,
                                 8, 9, 10, 11, 12, 13, 14, 15);
}

// B: 32x16 f16, B[k][n] = W[col0+n][kb+k]; lanes 0-15 hold K {0..7,16..23},
//    lanes 16-31 hold K {8..15,24..31} (same order as A).
__device__ __forceinline__ v16h load_frag_b_lds(const _Float16* sW,
                                                int col0, int kb, int lane) {
  const int n    = lane & 15;
  const int hi16 = (lane & 16) >> 1;
  const _Float16* p = sW + (col0 + n) * FDIM + kb + hi16;
  v8h lo = *reinterpret_cast<const v8h*>(p);
  v8h hh = *reinterpret_cast<const v8h*>(p + 16);
  return __builtin_shufflevector(lo, hh, 0, 1, 2, 3, 4, 5, 6, 7,
                                 8, 9, 10, 11, 12, 13, 14, 15);
}

// ---- fused (dual) GEMM tile kernel -----------------------------------------
// out[r, c] = act( sum_k A1[r,k] W1[c,k] (+ sum_k A2[r,k] W2[c,k]) + bias[c] )
// One wave per 16x16 tile; blockDim = (32, COLS/16); gridDim.x = N/16.
// Weights staged into LDS via TDM; shared A tile staged cooperatively.
template <bool DUAL, int ACT, int COLS>
__global__ void wmma_gemm_kernel(const _Float16* __restrict__ A1,
                                 const _Float16* __restrict__ W1,
                                 const _Float16* __restrict__ A2,
                                 const _Float16* __restrict__ W2,
                                 const float* __restrict__ bias,
                                 _Float16* __restrict__ out, int ld_out) {
  extern __shared__ char smem[];
  constexpr unsigned offW1 = 0;
  constexpr unsigned offW2 = (unsigned)COLS * FDIM * 2;
  constexpr unsigned offA1 = offW2 + (DUAL ? FDIM * FDIM * 2 : 0);
  constexpr unsigned offA2 = offA1 + 16 * FDIM * 2;
  _Float16* sW1 = reinterpret_cast<_Float16*>(smem + offW1);
  _Float16* sW2 = reinterpret_cast<_Float16*>(smem + offW2);
  _Float16* sA1 = reinterpret_cast<_Float16*>(smem + offA1);
  _Float16* sA2 = reinterpret_cast<_Float16*>(smem + offA2);

  const int lane = threadIdx.x;          // 0..31
  const int row0 = blockIdx.x * 16;
  const int col0 = threadIdx.y * 16;
  const int tid  = threadIdx.y * 32 + threadIdx.x;
  const int nthr = blockDim.y * 32;

  // ---- stage weights into LDS via the Tensor Data Mover (wave 0 issues) ----
#ifdef USE_TDM
  if (threadIdx.y == 0) {
    tdm_load_w(W1, offW1, COLS);
    if (DUAL) tdm_load_w(W2, offW2, FDIM);
    __builtin_amdgcn_s_wait_tensorcnt(0);
  }
#else
  for (int i = tid * 8; i < COLS * FDIM; i += nthr * 8)
    *reinterpret_cast<v8h*>(sW1 + i) = *reinterpret_cast<const v8h*>(W1 + i);
  if (DUAL)
    for (int i = tid * 8; i < FDIM * FDIM; i += nthr * 8)
      *reinterpret_cast<v8h*>(sW2 + i) = *reinterpret_cast<const v8h*>(W2 + i);
#endif

  // ---- stage the shared 16x128 A tile(s) cooperatively ---------------------
  for (int i = tid * 8; i < 16 * FDIM; i += nthr * 8) {
    *reinterpret_cast<v8h*>(sA1 + i) =
        *reinterpret_cast<const v8h*>(A1 + (size_t)row0 * FDIM + i);
    if (DUAL)
      *reinterpret_cast<v8h*>(sA2 + i) =
          *reinterpret_cast<const v8h*>(A2 + (size_t)row0 * FDIM + i);
  }
  __syncthreads();

  // ---- K loop: 4 x (one or two chained WMMAs into the same accumulator) ----
  v8f acc = {};
#pragma unroll
  for (int kb = 0; kb < FDIM; kb += 32) {
    v16h a1 = load_frag_a_lds(sA1, kb, lane);
    v16h b1 = load_frag_b_lds(sW1, col0, kb, lane);
    acc = __builtin_amdgcn_wmma_f32_16x16x32_f16(
        false, a1, false, b1, (short)0, acc, false, false);
    if (DUAL) {
      v16h a2 = load_frag_a_lds(sA2, kb, lane);
      v16h b2 = load_frag_b_lds(sW2, col0, kb, lane);
      acc = __builtin_amdgcn_wmma_f32_16x16x32_f16(
          false, a2, false, b2, (short)0, acc, false, false);
    }
  }

  // C/D layout: VGPR r -> row M = r (+8 for lanes 16-31); col N = lane&15.
  const int n     = lane & 15;
  const int mbase = row0 + ((lane & 16) >> 1);
  const float bsum = bias[col0 + n];
#pragma unroll
  for (int r = 0; r < 8; ++r) {
    float v = acc[r] + bsum;
    if (ACT == 1) v = fmaxf(v, 0.0f);
    out[(size_t)(mbase + r) * ld_out + col0 + n] = (_Float16)v;
  }
}

// ---- final head: [N,64] @ Wlin3^T(6x64) + b, sigmoid -----------------------
__global__ void head_kernel(const _Float16* __restrict__ h,
                            const float* __restrict__ W,
                            const float* __restrict__ b,
                            float* __restrict__ out, int N) {
  int n = blockIdx.x * blockDim.x + threadIdx.x;
  if (n >= N) return;
  const _Float16* hv = h + (size_t)n * 64;
#pragma unroll
  for (int o = 0; o < 6; ++o) {
    const float* w = W + o * 64;
    float s = b[o];
#pragma unroll 8
    for (int k = 0; k < 64; ++k) s += (float)hv[k] * w[k];
    out[(size_t)n * 6 + o] = 1.0f / (1.0f + __expf(-s));
  }
}

// ---------------------------------------------------------------------------
extern "C" void kernel_launch(void* const* d_in, const int* in_sizes, int n_in,
                              void* d_out, int out_size, void* d_ws, size_t ws_size,
                              hipStream_t stream) {
  const int N = in_sizes[0] / 3;   // 100000
  const int E = in_sizes[1] / 2;   // 1600000

  const float* x    = (const float*)d_in[0];
  const int*   ei   = (const int*)d_in[1];
  const int*   src  = ei;
  const int*   dst  = ei + E;
  const float* W1l  = (const float*)d_in[2];
  const float* b1l  = (const float*)d_in[3];
  const float* W1r  = (const float*)d_in[4];
  const float* W2l  = (const float*)d_in[5];
  const float* b2l  = (const float*)d_in[6];
  const float* W2r  = (const float*)d_in[7];
  const float* W3l  = (const float*)d_in[8];
  const float* b3l  = (const float*)d_in[9];
  const float* W3r  = (const float*)d_in[10];
  const float* Wl1  = (const float*)d_in[11];
  const float* bl1  = (const float*)d_in[12];
  const float* Wl2  = (const float*)d_in[13];
  const float* bl2  = (const float*)d_in[14];
  const float* Wl3  = (const float*)d_in[15];
  const float* bl3  = (const float*)d_in[16];

  // ---- workspace: agg f32 + f16 activation ping-pong + f16 weight slots ----
  float*    aggb = (float*)d_ws;                            // N*128 f32
  _Float16* h16a = (_Float16*)(aggb + (size_t)N * FDIM);    // N*128 f16
  _Float16* h16b = h16a + (size_t)N * FDIM;                 // N*128 f16
  _Float16* a16  = h16b + (size_t)N * FDIM;                 // N*128 f16 (agg)
  _Float16* wl16 = a16 + (size_t)N * FDIM;                  // 128*128 f16
  _Float16* wr16 = wl16 + FDIM * FDIM;                      // 128*128 f16

  const int T  = 256;
  const int nf = N * FDIM;
  const int n3 = N * 3;
  const int gridScat = (int)(((size_t)E * 32 + T - 1) / T);
  const int gW = (FDIM * FDIM + T - 1) / T;
  dim3 gBlk(N / 16);
  dim3 tBlk8(32, 8);   // 128 output cols
  dim3 tBlk4(32, 4);   // 64 output cols
  const size_t shDual = (size_t)(FDIM * FDIM * 2 + 16 * FDIM * 2) * sizeof(_Float16);
  const size_t shS128 = (size_t)(FDIM * FDIM + 16 * FDIM) * sizeof(_Float16);
  const size_t shS64  = (size_t)(64 * FDIM + 16 * FDIM) * sizeof(_Float16);

  // ---- layer 1: 3 -> 128 ----
  fill_neginf_kernel<<<(n3 + T - 1) / T, T, 0, stream>>>((unsigned int*)aggb, n3);
  scatter_max3_kernel<<<(E + T - 1) / T, T, 0, stream>>>(x, src, dst, aggb, E);
  layer1_kernel<<<(nf + T - 1) / T, T, 0, stream>>>(x, aggb, W1l, b1l, W1r, h16a, N);

  // ---- layer 2: 128 -> 128 ----
  pack_w16_kernel<<<gW, T, 0, stream>>>(W2l, wl16, FDIM * FDIM);
  pack_w16_kernel<<<gW, T, 0, stream>>>(W2r, wr16, FDIM * FDIM);
  fill_neginf_kernel<<<(nf + T - 1) / T, T, 0, stream>>>((unsigned int*)aggb, nf);
  scatter_max128_kernel<<<gridScat, T, 0, stream>>>(h16a, src, dst, aggb, E);
  pack_agg16_kernel<<<(nf + T - 1) / T, T, 0, stream>>>(aggb, a16, nf);
  wmma_gemm_kernel<true, 0, FDIM><<<gBlk, tBlk8, shDual, stream>>>(
      a16, wl16, h16a, wr16, b2l, h16b, FDIM);

  // ---- layer 3: 128 -> 128 ----
  pack_w16_kernel<<<gW, T, 0, stream>>>(W3l, wl16, FDIM * FDIM);
  pack_w16_kernel<<<gW, T, 0, stream>>>(W3r, wr16, FDIM * FDIM);
  fill_neginf_kernel<<<(nf + T - 1) / T, T, 0, stream>>>((unsigned int*)aggb, nf);
  scatter_max128_kernel<<<gridScat, T, 0, stream>>>(h16b, src, dst, aggb, E);
  pack_agg16_kernel<<<(nf + T - 1) / T, T, 0, stream>>>(aggb, a16, nf);
  wmma_gemm_kernel<true, 0, FDIM><<<gBlk, tBlk8, shDual, stream>>>(
      a16, wl16, h16b, wr16, b3l, h16a, FDIM);

  // ---- MLP head ----
  pack_w16_kernel<<<gW, T, 0, stream>>>(Wl1, wl16, FDIM * FDIM);
  wmma_gemm_kernel<false, 1, FDIM><<<gBlk, tBlk8, shS128, stream>>>(
      h16a, wl16, nullptr, nullptr, bl1, h16b, FDIM);
  pack_w16_kernel<<<(64 * FDIM + T - 1) / T, T, 0, stream>>>(Wl2, wl16, 64 * FDIM);
  wmma_gemm_kernel<false, 1, 64><<<gBlk, tBlk4, shS64, stream>>>(
      h16b, wl16, nullptr, nullptr, bl2, h16a, 64);
  head_kernel<<<(N + T - 1) / T, T, 0, stream>>>(h16a, Wl3, bl3, (float*)d_out, N);
}
